// MultiHeadAtention_84327387890467
// MI455X (gfx1250) — matrix operand order repaired and checked by reading, not compiled, not run on hardware
//
#include <hip/hip_runtime.h>
#include <hip/hip_bf16.h>

// ---------------------------------------------------------------------------
// CDNA5 (gfx1250) multi-head attention, bf16 WMMA pipeline, wave32.
//   D_MODEL=1024, H=16, d_k=64, B=4, S=2048  (M = B*S = 8192)
// Double-buffered async global->LDS staging; DPP16 softmax reductions.
// ---------------------------------------------------------------------------

#define USE_ASYNC_LDS 1   // global_load_async_to_lds_b128 + s_wait_asynccnt

typedef __bf16 v16bf __attribute__((ext_vector_type(16)));
typedef float  v8f   __attribute__((ext_vector_type(8)));

union BFrag { v16bf v; uint4 q[2]; };

__device__ __forceinline__ unsigned short f2bf(float f) {
  unsigned int u = __float_as_uint(f);
  u += 0x7fffu + ((u >> 16) & 1u);            // round-to-nearest-even
  return (unsigned short)(u >> 16);
}

// packed fp32x2 -> bf16x2
__device__ __forceinline__ unsigned pk_bf16(float a, float b) {
#if __has_builtin(__builtin_amdgcn_cvt_pk_bf16_f32)
  typedef __bf16 v2bf __attribute__((ext_vector_type(2)));
  union { v2bf v; unsigned u; } cv;
  cv.v = __builtin_amdgcn_cvt_pk_bf16_f32(a, b);
  return cv.u;
#else
  unsigned ua = __float_as_uint(a), ub = __float_as_uint(b);
  ua += 0x7fffu + ((ua >> 16) & 1u);
  ub += 0x7fffu + ((ub >> 16) & 1u);
#if __has_builtin(__builtin_amdgcn_perm)
  return __builtin_amdgcn_perm(ub, ua, 0x07060302u);   // {ub.hi, ua.hi}
#else
  return (ub & 0xffff0000u) | (ua >> 16);
#endif
#endif
}

__device__ __forceinline__ v8f wmma_bf16(v16bf a, v16bf b, v8f c) {
  return __builtin_amdgcn_wmma_f32_16x16x32_bf16(
      false, a, false, b, (short)0, c, false, false);
}

// ---- 16-lane-row reductions via DPP16 (no LDS traffic) ----
#if __has_builtin(__builtin_amdgcn_update_dpp)
#define DPPF(v, ctrl) \
  __int_as_float(__builtin_amdgcn_update_dpp(0, __float_as_int(v), ctrl, 0xf, 0xf, true))
__device__ __forceinline__ float row16_max(float v) {
  v = fmaxf(v, DPPF(v, 0xB1));    // quad_perm 1,0,3,2  (xor 1)
  v = fmaxf(v, DPPF(v, 0x4E));    // quad_perm 2,3,0,1  (xor 2)
  v = fmaxf(v, DPPF(v, 0x141));   // row_half_mirror
  v = fmaxf(v, DPPF(v, 0x140));   // row_mirror
  return v;
}
__device__ __forceinline__ float row16_sum(float v) {
  v += DPPF(v, 0xB1);
  v += DPPF(v, 0x4E);
  v += DPPF(v, 0x141);
  v += DPPF(v, 0x140);
  return v;
}
#else
__device__ __forceinline__ float row16_max(float v) {
  for (int o = 1; o < 16; o <<= 1) v = fmaxf(v, __shfl_xor(v, o, 32));
  return v;
}
__device__ __forceinline__ float row16_sum(float v) {
  for (int o = 1; o < 16; o <<= 1) v += __shfl_xor(v, o, 32);
  return v;
}
#endif

// ---- CDNA5 async global -> LDS staging (ASYNCcnt-tracked) ----
__device__ __forceinline__ void async_ld_lds_32B(void* lds, const void* gmem) {
#if USE_ASYNC_LDS
  unsigned l = (unsigned)(size_t)lds;    // low 32 bits = LDS byte offset
  asm volatile("global_load_async_to_lds_b128 %0, %1, off\n\t"
               "global_load_async_to_lds_b128 %0, %1, off offset:16"
               :: "v"(l), "v"(gmem) : "memory");
#else
  ((uint4*)lds)[0] = ((const uint4*)gmem)[0];
  ((uint4*)lds)[1] = ((const uint4*)gmem)[1];
#endif
}
// wait until at most N async ops outstanding (N = group just issued)
template <int N>
__device__ __forceinline__ void wait_async_le() {
#if USE_ASYNC_LDS
  asm volatile("s_wait_asynccnt %0" :: "i"(N) : "memory");
#endif
}

// ---------------------------------------------------------------------------
// One-time fp32 -> bf16 weight conversion (each thread: 4 floats)
// ---------------------------------------------------------------------------
__global__ __launch_bounds__(256) void cvt_f32_bf16(
    const float* __restrict__ src, unsigned short* __restrict__ dst)
{
  const int i = blockIdx.x * blockDim.x + threadIdx.x;
  const float4 f = ((const float4*)src)[i];
  ((uint2*)dst)[i] = make_uint2(pk_bf16(f.x, f.y), pk_bf16(f.z, f.w));
}

// ---------------------------------------------------------------------------
// Projection GEMM:  Y[m][n] = (sum_k X[m][k] * W[n][k] + bias[n]) * scale
// Block tile 128x64, 4 waves (32 M-rows each), K-step 32, double-buffered LDS.
// MODE 0: bf16 out [B][H][S][64];  MODE 1: bf16 out [B][H][64][S] (V^T);
// MODE 2: fp32 out flat [M][1024].   XBF: X is bf16 (else fp32, fused cvt).
// ---------------------------------------------------------------------------
template <int MODE, bool XBF>
__global__ __launch_bounds__(128) void mha_proj_gemm(
    const float* __restrict__ Xf, const unsigned short* __restrict__ Xb,
    const unsigned short* __restrict__ Wb, const float* __restrict__ bias,
    void* __restrict__ Y, float out_scale)
{
  __shared__ __align__(16) unsigned short sA[2][128 * 32];
  __shared__ __align__(16) unsigned short sB[2][64 * 32];

  const int t    = threadIdx.x;
  const int wave = t >> 5;
  const int lane = t & 31;
  const int hl   = lane >> 4;
  const int ln   = lane & 15;
  const int m0   = blockIdx.y * 128;
  const int n0   = blockIdx.x * 64;
  const int brow = t >> 1, bcol = (t & 1) * 16;   // B staging: 64 x 32

  constexpr int AG = XBF ? 6 : 2;   // async instructions issued per stage

  auto stage = [&](int k0, int buf) {
    async_ld_lds_32B(sB[buf] + brow * 32 + bcol,
                     Wb + (size_t)(n0 + brow) * 1024 + k0 + bcol);
    if constexpr (XBF) {            // A row (64B) via async DMA
      const unsigned short* xr = Xb + (size_t)(m0 + t) * 1024 + k0;
      async_ld_lds_32B(sA[buf] + t * 32, xr);
      async_ld_lds_32B(sA[buf] + t * 32 + 16, xr + 16);
    } else {                        // A row: fp32 load + packed bf16 convert
      const float4* s = (const float4*)(Xf + (size_t)(m0 + t) * 1024 + k0);
      uint4* d = (uint4*)(sA[buf] + t * 32);
      #pragma unroll
      for (int i = 0; i < 4; ++i) {
        const float4 fa = s[2 * i], fb = s[2 * i + 1];
        d[i] = make_uint4(pk_bf16(fa.x, fa.y), pk_bf16(fa.z, fa.w),
                          pk_bf16(fb.x, fb.y), pk_bf16(fb.z, fb.w));
      }
    }
  };

  v8f acc[2][4] = {};
  stage(0, 0);
  int ib = 0;
  for (int k0 = 0; k0 < 1024; k0 += 32, ib ^= 1) {
    if (k0 + 32 < 1024) { stage(k0 + 32, ib ^ 1); wait_async_le<AG>(); }
    else                { wait_async_le<0>(); }
    __syncthreads();

    // two A fragments (rows wave*32 .. +31)
    BFrag a0, a1;
    const unsigned short* ap0 = sA[ib] + (wave * 32 + ln) * 32;
    const unsigned short* ap1 = ap0 + 16 * 32;
    a0.q[0] = *(const uint4*)(ap0 + hl * 8);
    a0.q[1] = *(const uint4*)(ap0 + 16 + hl * 8);
    a1.q[0] = *(const uint4*)(ap1 + hl * 8);
    a1.q[1] = *(const uint4*)(ap1 + 16 + hl * 8);

    #pragma unroll
    for (int nt = 0; nt < 4; ++nt) {
      BFrag b;
      const unsigned short* bp = sB[ib] + (nt * 16 + ln) * 32 + hl * 8;
      b.q[0] = *(const uint4*)bp;
      b.q[1] = *(const uint4*)(bp + 16);
      acc[0][nt] = wmma_bf16(a0.v, b.v, acc[0][nt]);
      acc[1][nt] = wmma_bf16(a1.v, b.v, acc[1][nt]);
    }
    __syncthreads();
  }

  // ---- epilogue: bias, scale, store ----
  #pragma unroll
  for (int mi = 0; mi < 2; ++mi) {
    #pragma unroll
    for (int nt = 0; nt < 4; ++nt) {
      const int n  = n0 + nt * 16 + ln;
      const float bv = bias[n];
      if constexpr (MODE == 0) {             // bf16 [B][H][S][64]
        unsigned short* Yp = (unsigned short*)Y;
        const int h = n >> 6, d = n & 63;
        #pragma unroll
        for (int r = 0; r < 8; ++r) {
          const int m = m0 + wave * 32 + mi * 16 + hl * 8 + r;
          const int bb = m >> 11, s = m & 2047;
          Yp[(((size_t)(bb * 16 + h) * 2048 + s) << 6) + d] =
              f2bf((acc[mi][nt][r] + bv) * out_scale);
        }
      } else if constexpr (MODE == 1) {      // bf16 [B][H][64][S]  (V^T)
        unsigned short* Yp = (unsigned short*)Y;
        const int h = n >> 6, d = n & 63;
        const int mb = m0 + wave * 32 + mi * 16 + hl * 8;
        const int bb = mb >> 11, s = mb & 2047;
        union { uint4 q; unsigned u[4]; } p;
        #pragma unroll
        for (int r = 0; r < 4; ++r)
          p.u[r] = pk_bf16((acc[mi][nt][2 * r] + bv) * out_scale,
                           (acc[mi][nt][2 * r + 1] + bv) * out_scale);
        *(uint4*)(Yp + ((size_t)(bb * 16 + h) * 64 + d) * 2048 + s) = p.q;
      } else {                               // fp32 flat [M][1024]
        float* Yp = (float*)Y;
        #pragma unroll
        for (int r = 0; r < 8; ++r) {
          const int m = m0 + wave * 32 + mi * 16 + hl * 8 + r;
          Yp[(size_t)m * 1024 + n] = (acc[mi][nt][r] + bv) * out_scale;
        }
      }
    }
  }
}

// ---------------------------------------------------------------------------
// Flash attention: one wave = 16 query rows x d_k=64 output; 4 waves share
// (b,h) and double-buffered K/V tiles (kv step 64); online softmax with
// DPP16 row reductions.  Q pre-scaled by 1/sqrt(d_k).
// ---------------------------------------------------------------------------
__global__ __launch_bounds__(128) void mha_attn(
    const unsigned short* __restrict__ Qp,   // bf16 [BH][2048][64]
    const unsigned short* __restrict__ Kp,   // bf16 [BH][2048][64]
    const unsigned short* __restrict__ Vt,   // bf16 [BH][64][2048]
    unsigned short* __restrict__ Ctx)        // bf16 [B*S][1024]
{
  __shared__ __align__(16) unsigned short sK[2][64 * 64];    // [kv][d]
  __shared__ __align__(16) unsigned short sV[2][64 * 64];    // [d][kv]
  __shared__ __align__(16) unsigned short sP[4][16 * 64];    // per-wave P

  const int t    = threadIdx.x;
  const int wave = t >> 5;
  const int lane = t & 31;
  const int hl   = lane >> 4;
  const int ln   = lane & 15;
  const int bh   = blockIdx.y;                 // b*16 + h
  const int m0   = blockIdx.x * 64 + wave * 16;

  const unsigned short* Qb = Qp + (size_t)bh * 2048 * 64;
  const unsigned short* Kb = Kp + (size_t)bh * 2048 * 64;
  const unsigned short* Vb = Vt + (size_t)bh * 64 * 2048;

  // Q A-fragments (k=0..31, 32..63) kept in registers
  BFrag qa0, qa1;
  {
    const unsigned short* qr = Qb + (size_t)(m0 + ln) * 64;
    qa0.q[0] = *(const uint4*)(qr + hl * 8);
    qa0.q[1] = *(const uint4*)(qr + 16 + hl * 8);
    qa1.q[0] = *(const uint4*)(qr + 32 + hl * 8);
    qa1.q[1] = *(const uint4*)(qr + 48 + hl * 8);
  }

  v8f acc[4] = {};
  float mrow[8], lrow[8];
  #pragma unroll
  for (int r = 0; r < 8; ++r) { mrow[r] = -1e30f; lrow[r] = 0.f; }

  const int srow = t >> 1, scol = (t & 1) * 32;   // staging: 64 rows x 64 cols

  auto stageKV = [&](int kv0, int buf) {
    const unsigned short* kg = Kb + (size_t)(kv0 + srow) * 64 + scol;
    async_ld_lds_32B(sK[buf] + srow * 64 + scol, kg);
    async_ld_lds_32B(sK[buf] + srow * 64 + scol + 16, kg + 16);
    const unsigned short* vg = Vb + (size_t)srow * 2048 + kv0 + scol;
    async_ld_lds_32B(sV[buf] + srow * 64 + scol, vg);
    async_ld_lds_32B(sV[buf] + srow * 64 + scol + 16, vg + 16);
  };

  stageKV(0, 0);
  int ib = 0;
  for (int kv0 = 0; kv0 < 2048; kv0 += 64, ib ^= 1) {
    if (kv0 + 64 < 2048) { stageKV(kv0 + 64, ib ^ 1); wait_async_le<8>(); }
    else                 { wait_async_le<0>(); }
    __syncthreads();

    // ---- S = Q * K^T : 16 x 64 as four 16x16 C fragments ----
    v8f sc[4];
    #pragma unroll
    for (int j = 0; j < 4; ++j) {
      BFrag b0, b1;
      const unsigned short* kp = sK[ib] + (j * 16 + ln) * 64 + hl * 8;
      b0.q[0] = *(const uint4*)kp;        b0.q[1] = *(const uint4*)(kp + 16);
      b1.q[0] = *(const uint4*)(kp + 32); b1.q[1] = *(const uint4*)(kp + 48);
      v8f z = {};
      z = wmma_bf16(qa0.v, b0.v, z);
      sc[j] = wmma_bf16(qa1.v, b1.v, z);
    }

    // ---- online softmax (C layout: row M = r + 8*hl, N = lane%16) ----
    float al[8];
    #pragma unroll
    for (int r = 0; r < 8; ++r) {
      float mx = fmaxf(fmaxf(sc[0][r], sc[1][r]), fmaxf(sc[2][r], sc[3][r]));
      mx = row16_max(mx);
      const float mn = fmaxf(mrow[r], mx);
      al[r] = __expf(mrow[r] - mn);
      float rs = 0.f;
      #pragma unroll
      for (int j = 0; j < 4; ++j) {
        const float p = __expf(sc[j][r] - mn);
        sc[j][r] = p;
        rs += p;
      }
      rs = row16_sum(rs);
      lrow[r] = lrow[r] * al[r] + rs;
      mrow[r] = mn;
    }
    #pragma unroll
    for (int nt = 0; nt < 4; ++nt)
      #pragma unroll
      for (int r = 0; r < 8; ++r) acc[nt][r] *= al[r];

    // ---- C-layout -> A-layout relayout of P via per-wave LDS tile ----
    unsigned short* Pw = sP[wave];
    #pragma unroll
    for (int r = 0; r < 8; ++r) {
      const int m = hl * 8 + r;
      #pragma unroll
      for (int j = 0; j < 4; ++j)
        Pw[m * 64 + j * 16 + ln] = f2bf(sc[j][r]);
    }
    BFrag pa0, pa1;  // same-wave DS in-order; wave-private region -> no barrier
    {
      const unsigned short* pp = Pw + ln * 64;
      pa0.q[0] = *(const uint4*)(pp + hl * 8);
      pa0.q[1] = *(const uint4*)(pp + 16 + hl * 8);
      pa1.q[0] = *(const uint4*)(pp + 32 + hl * 8);
      pa1.q[1] = *(const uint4*)(pp + 48 + hl * 8);
    }

    // ---- O += P * V  (kv chunks 0..31, 32..63) ----
    #pragma unroll
    for (int nt = 0; nt < 4; ++nt) {
      BFrag v0, v1;
      const unsigned short* vp = sV[ib] + (nt * 16 + ln) * 64 + hl * 8;
      v0.q[0] = *(const uint4*)vp;        v0.q[1] = *(const uint4*)(vp + 16);
      v1.q[0] = *(const uint4*)(vp + 32); v1.q[1] = *(const uint4*)(vp + 48);
      acc[nt] = wmma_bf16(pa0.v, v0.v, acc[nt]);
      acc[nt] = wmma_bf16(pa1.v, v1.v, acc[nt]);
    }
    __syncthreads();
  }

  // ---- finalize: divide by row sums, write context bf16 [B*S][1024] ----
  const int h  = bh & 15;
  const int bb = bh >> 4;
  float inv[8];
  #pragma unroll
  for (int r = 0; r < 8; ++r) inv[r] = 1.f / lrow[r];
  #pragma unroll
  for (int nt = 0; nt < 4; ++nt) {
    const int n = h * 64 + nt * 16 + ln;
    #pragma unroll
    for (int r = 0; r < 8; ++r) {
      const int s = m0 + hl * 8 + r;
      Ctx[((size_t)(bb * 2048 + s)) * 1024 + n] = f2bf(acc[nt][r] * inv[r]);
    }
  }
}

// ---------------------------------------------------------------------------
extern "C" void kernel_launch(void* const* d_in, const int* in_sizes, int n_in,
                              void* d_out, int out_size, void* d_ws, size_t ws_size,
                              hipStream_t stream) {
  (void)in_sizes; (void)n_in; (void)out_size; (void)ws_size;
  const float* query = (const float*)d_in[0];
  const float* key_  = (const float*)d_in[1];
  const float* value = (const float*)d_in[2];
  const float* wq    = (const float*)d_in[3];
  const float* bq    = (const float*)d_in[4];
  const float* wk    = (const float*)d_in[5];
  const float* bk    = (const float*)d_in[6];
  const float* wo    = (const float*)d_in[7];
  const float* bo    = (const float*)d_in[8];

  char* ws = (char*)d_ws;
  const size_t MB = (size_t)1024 * 1024;
  unsigned short* Qp  = (unsigned short*)(ws);             // 16 MB each
  unsigned short* Kp  = (unsigned short*)(ws + 16 * MB);
  unsigned short* Vt  = (unsigned short*)(ws + 32 * MB);
  unsigned short* Ctx = (unsigned short*)(ws + 48 * MB);
  unsigned short* Wqb = (unsigned short*)(ws + 64 * MB);   // 2 MB each
  unsigned short* Wkb = (unsigned short*)(ws + 66 * MB);
  unsigned short* Wob = (unsigned short*)(ws + 68 * MB);

  const dim3 blk(128);
  const dim3 gcvt(1024);       // 1024*256 threads * 4 floats = 1M elements
  const dim3 gproj(16, 64);    // N-tiles x M-tiles (128x64 tiles over 8192x1024)
  const dim3 gattn(32, 64);    // S/64 query groups x (B*H)

  // one-time weight conversion fp32 -> bf16
  cvt_f32_bf16<<<gcvt, 256, 0, stream>>>(wq, Wqb);
  cvt_f32_bf16<<<gcvt, 256, 0, stream>>>(wk, Wkb);
  cvt_f32_bf16<<<gcvt, 256, 0, stream>>>(wo, Wob);

  // Q = (query Wq^T + bq)/8;  K = key Wk^T + bk;  V = value Wk^T + bk (ref quirk)
  mha_proj_gemm<0, false><<<gproj, blk, 0, stream>>>(query, nullptr, Wqb, bq, Qp, 0.125f);
  mha_proj_gemm<0, false><<<gproj, blk, 0, stream>>>(key_,  nullptr, Wkb, bk, Kp, 1.0f);
  mha_proj_gemm<1, false><<<gproj, blk, 0, stream>>>(value, nullptr, Wkb, bk, Vt, 1.0f);
  mha_attn<<<gattn, blk, 0, stream>>>(Qp, Kp, Vt, Ctx);
  mha_proj_gemm<2, true><<<gproj, blk, 0, stream>>>(nullptr, Ctx, Wob, bo, d_out, 1.0f);
}
